// MambaBlock_69226282877355
// MI455X (gfx1250) — compile-verified, matrix operand order checked by reading, not run on hardware
//
#include <hip/hip_runtime.h>
#include <math.h>

// MI455X / gfx1250, wave32. fp16 WMMA (16x16x32) with f32 accumulate.
typedef _Float16 v16h __attribute__((ext_vector_type(16)));
typedef _Float16 v4h  __attribute__((ext_vector_type(4)));
typedef float    v8f  __attribute__((ext_vector_type(8)));

#define D_MODEL 1024
#define D_STATE 16
#define D_CONV  4
#define D_INNER 2048
#define DT_RANK 64
#define BATCH   2
#define SEQ_L   2048
#define NTOK    (BATCH * SEQ_L)   // 4096 tokens
#define DBC_LD  128               // padded row stride for dbc / W_xproj

#define LDS_STRIDE 48             // halves per row: 96B -> 32B-aligned fragments

__device__ __forceinline__ float siluf(float x) { return x / (1.0f + expf(-x)); }
__device__ __forceinline__ float softplusf(float x) {
    return fmaxf(x, 0.0f) + log1pf(expf(-fabsf(x)));
}

// ---------------------------------------------------------------------------
// LayerNorm: one block per token, 256 threads, single-pass sum/sumsq reduce.
// ---------------------------------------------------------------------------
__global__ __launch_bounds__(256)
void ln_kernel(const float* __restrict__ x, const float* __restrict__ w,
               const float* __restrict__ b, float* __restrict__ out) {
    __shared__ float s1[256], s2[256];
    const int t = blockIdx.x;
    const float* xr = x + (size_t)t * D_MODEL;
    float a0 = 0.f, a1 = 0.f;
    for (int i = threadIdx.x; i < D_MODEL; i += 256) {
        float v = xr[i]; a0 += v; a1 += v * v;
    }
    s1[threadIdx.x] = a0; s2[threadIdx.x] = a1;
    __syncthreads();
    for (int o = 128; o > 0; o >>= 1) {
        if (threadIdx.x < o) {
            s1[threadIdx.x] += s1[threadIdx.x + o];
            s2[threadIdx.x] += s2[threadIdx.x + o];
        }
        __syncthreads();
    }
    const float mean = s1[0] * (1.0f / D_MODEL);
    const float var  = s2[0] * (1.0f / D_MODEL) - mean * mean;
    const float rstd = rsqrtf(var + 1e-5f);
    float* orow = out + (size_t)t * D_MODEL;
    for (int i = threadIdx.x; i < D_MODEL; i += 256)
        orow[i] = (xr[i] - mean) * rstd * w[i] + b[i];
}

// ---------------------------------------------------------------------------
// Zero-pad W_xproj (2048 x 96) into (2048 x 128) so the dbc GEMM is guard-free.
// ---------------------------------------------------------------------------
__global__ __launch_bounds__(256)
void pad_wxproj_kernel(const float* __restrict__ w, float* __restrict__ wp) {
    const int idx = blockIdx.x * 256 + threadIdx.x;   // D_INNER*128
    const int r = idx >> 7, c = idx & 127;
    wp[idx] = (c < 96) ? w[r * 96 + c] : 0.f;
}

// ---------------------------------------------------------------------------
// WMMA GEMM: C(MxN,f32) = A(MxK,f32,rm,lda) * B(KxN,f32,rm,ldb)
// Requirements: M%128==0, N%128==0, K%32==0 (all call sites satisfy this).
// Block tile 128x128, 8 waves (4 M x 2 N), wave tile 32x64 = 2x4 WMMA tiles.
// f32->f16 conversion during LDS staging; LDS is stored K-permuted so each
// lane's 16-half fragment is contiguous (2x ds_load_b128). Double-buffered.
// MODE 0: C = acc
// MODE 1: C = softplus(acc + bias[col])
// MODE 2: C = acc + resid[row*ldr + col]
// ---------------------------------------------------------------------------
template<int MODE>
__global__ __launch_bounds__(256)
void wmma_gemm(const float* __restrict__ A, int lda,
               const float* __restrict__ Bg, int ldb,
               float* __restrict__ C, int ldc,
               int M, int N, int K,
               const float* __restrict__ bias,
               const float* __restrict__ resid, int ldr) {
    __shared__ _Float16 As[2][128 * LDS_STRIDE];   // 2 x 12 KB
    __shared__ _Float16 Bt[2][128 * LDS_STRIDE];   // 2 x 12 KB (col-major: [n][k])

    const int tid  = threadIdx.x;
    const int lane = tid & 31;
    const int wid  = tid >> 5;
    const int wm   = wid & 3;            // 4 waves * 32 rows = 128
    const int wn   = wid >> 2;           // 2 waves * 64 cols = 128
    const int lo   = lane & 15;
    const int hi   = lane >> 4;
    const size_t bm = (size_t)blockIdx.y * 128;
    const size_t bn = (size_t)blockIdx.x * 128;

    float4 ra[4], rb[4];

    auto load_tile = [&](int kk) {
        #pragma unroll
        for (int i = 0; i < 4; ++i) {              // A: 128x32 = 1024 float4
            int e = tid + 256 * i;
            int r = e >> 3, c4 = (e & 7) << 2;
            ra[i] = *reinterpret_cast<const float4*>(&A[(bm + r) * (size_t)lda + kk + c4]);
        }
        #pragma unroll
        for (int i = 0; i < 4; ++i) {              // B: 32x128 = 1024 float4
            int e = tid + 256 * i;
            int k = e >> 5, c4 = (e & 31) << 2;
            rb[i] = *reinterpret_cast<const float4*>(&Bg[(size_t)(kk + k) * ldb + bn + c4]);
        }
    };

    // K-permutation p(K) = (K&7) + 16*bit3(K) + 8*bit4(K): makes the WMMA
    // fragment (half j of lane l -> K=(j&7)+8*(l>=16)+16*(j>>3)) contiguous.
    auto store_tile = [&](int buf) {
        #pragma unroll
        for (int i = 0; i < 4; ++i) {
            int e = tid + 256 * i;
            int r = e >> 3, c4 = (e & 7) << 2;
            int p = (c4 & 7) + (((c4 >> 3) & 1) << 4) + (((c4 >> 4) & 1) << 3);
            v4h h = { (_Float16)ra[i].x, (_Float16)ra[i].y,
                      (_Float16)ra[i].z, (_Float16)ra[i].w };
            *reinterpret_cast<v4h*>(&As[buf][r * LDS_STRIDE + p]) = h;
        }
        #pragma unroll
        for (int i = 0; i < 4; ++i) {
            int e = tid + 256 * i;
            int k = e >> 5, c4 = (e & 31) << 2;
            int pk = (k & 7) + (((k >> 3) & 1) << 4) + (((k >> 4) & 1) << 3);
            const float* f = &rb[i].x;
            #pragma unroll
            for (int q = 0; q < 4; ++q)            // transpose scatter into [n][k]
                Bt[buf][(c4 + q) * LDS_STRIDE + pk] = (_Float16)f[q];
        }
    };

    const v8f vz = {0.f,0.f,0.f,0.f,0.f,0.f,0.f,0.f};
    v8f acc[2][4];
    #pragma unroll
    for (int i = 0; i < 2; ++i)
        #pragma unroll
        for (int j = 0; j < 4; ++j) acc[i][j] = vz;

    load_tile(0);
    store_tile(0);
    const int nk = K >> 5;
    int buf = 0;
    for (int kt = 0; kt < nk; ++kt) {
        const bool more = (kt + 1 < nk);
        if (more) load_tile((kt + 1) << 5);        // overlap HBM with WMMA
        __syncthreads();

        v16h af[2], bf[4];
        #pragma unroll
        for (int s = 0; s < 2; ++s)
            af[s] = *reinterpret_cast<const v16h*>(
                &As[buf][(wm * 32 + s * 16 + lo) * LDS_STRIDE + hi * 16]);
        #pragma unroll
        for (int s = 0; s < 4; ++s)
            bf[s] = *reinterpret_cast<const v16h*>(
                &Bt[buf][(wn * 64 + s * 16 + lo) * LDS_STRIDE + hi * 16]);

        #pragma unroll
        for (int i = 0; i < 2; ++i)
            #pragma unroll
            for (int j = 0; j < 4; ++j)
                acc[i][j] = __builtin_amdgcn_wmma_f32_16x16x32_f16(
                    false, af[i], false, bf[j], (short)0, acc[i][j], false, false);

        if (more) store_tile(buf ^ 1);
        buf ^= 1;
    }

    // C/D layout: VGPR r -> M = r + 8*(lane>=16); N = lane%16.
    #pragma unroll
    for (int i = 0; i < 2; ++i) {
        #pragma unroll
        for (int j = 0; j < 4; ++j) {
            #pragma unroll
            for (int r = 0; r < 8; ++r) {
                size_t row = bm + wm * 32 + i * 16 + r + 8 * hi;
                size_t col = bn + wn * 64 + j * 16 + lo;
                float v = acc[i][j][r];
                if (MODE == 1)      v = softplusf(v + bias[col]);
                else if (MODE == 2) v += resid[row * (size_t)ldr + col];
                C[row * (size_t)ldc + col] = v;
            }
        }
    }
}

// ---------------------------------------------------------------------------
// Depthwise causal conv (D_CONV=4) over L within each batch, + bias + SiLU.
// u lives in the first D_INNER columns of xz (row stride 2*D_INNER).
// ---------------------------------------------------------------------------
__global__ __launch_bounds__(256)
void conv_silu_kernel(const float* __restrict__ xz, const float* __restrict__ cw,
                      const float* __restrict__ cb, float* __restrict__ uact) {
    const int idx = blockIdx.x * 256 + threadIdx.x;   // over NTOK*D_INNER
    const int d  = idx & (D_INNER - 1);
    const int t  = idx >> 11;                          // global token
    const int tl = t & (SEQ_L - 1);                    // token within batch
    float acc = cb[d];
    #pragma unroll
    for (int k = 0; k < D_CONV; ++k) {
        int ts = tl - (D_CONV - 1) + k;
        if (ts >= 0) {
            float uv = xz[(size_t)(t - (D_CONV - 1) + k) * (2 * D_INNER) + d];
            acc += cw[d * D_CONV + k] * uv;
        }
    }
    uact[idx] = siluf(acc);
}

// ---------------------------------------------------------------------------
// Selective-scan: one thread per (b,d) channel; 16 states in registers;
// sequential over L. Fuses  y = (scan + u*D) * silu(z).
// ---------------------------------------------------------------------------
__global__ __launch_bounds__(256)
void scan_kernel(const float* __restrict__ delta, const float* __restrict__ uact,
                 const float* __restrict__ dbc,   const float* __restrict__ xz,
                 const float* __restrict__ A_log, const float* __restrict__ Dp,
                 float* __restrict__ ybuf) {
    const int idx = blockIdx.x * 256 + threadIdx.x;    // BATCH*D_INNER
    const int b = idx >> 11;
    const int d = idx & (D_INNER - 1);
    float Arow[D_STATE];
    #pragma unroll
    for (int s = 0; s < D_STATE; ++s) Arow[s] = -expf(A_log[d * D_STATE + s]);
    const float Dd = Dp[d];
    float h[D_STATE];
    #pragma unroll
    for (int s = 0; s < D_STATE; ++s) h[s] = 0.f;

    for (int t = 0; t < SEQ_L; ++t) {
        const size_t row = (size_t)b * SEQ_L + t;
        const float dt = delta[row * D_INNER + d];
        const float ut = uact[row * D_INNER + d];
        const float du = dt * ut;
        const float* bc = dbc + row * DBC_LD;          // [dt(64) | B(16) | C(16) | pad]
        float y = 0.f;
        #pragma unroll
        for (int s = 0; s < D_STATE; ++s) {
            float hv = expf(dt * Arow[s]) * h[s] + du * bc[DT_RANK + s];
            h[s] = hv;
            y += hv * bc[DT_RANK + D_STATE + s];
        }
        const float z = xz[row * (2 * D_INNER) + D_INNER + d];
        ybuf[row * D_INNER + d] = (y + ut * Dd) * siluf(z);
    }
}

// ---------------------------------------------------------------------------
extern "C" void kernel_launch(void* const* d_in, const int* in_sizes, int n_in,
                              void* d_out, int out_size, void* d_ws, size_t ws_size,
                              hipStream_t stream) {
    (void)in_sizes; (void)n_in; (void)out_size; (void)ws_size;
    const float* x      = (const float*)d_in[0];
    const float* ln_w   = (const float*)d_in[1];
    const float* ln_b   = (const float*)d_in[2];
    const float* W_in   = (const float*)d_in[3];
    const float* conv_w = (const float*)d_in[4];
    const float* conv_b = (const float*)d_in[5];
    const float* W_xprj = (const float*)d_in[6];
    const float* W_dt   = (const float*)d_in[7];
    const float* b_dt   = (const float*)d_in[8];
    const float* A_log  = (const float*)d_in[9];
    const float* Dp     = (const float*)d_in[10];
    const float* W_out  = (const float*)d_in[11];
    float* out = (float*)d_out;

    // Workspace layout (floats): ~188 MB total.
    float* ws = (float*)d_ws;
    size_t o = 0;
    float* normed  = ws + o; o += (size_t)NTOK * D_MODEL;        // 4096x1024
    float* xz      = ws + o; o += (size_t)NTOK * 2 * D_INNER;    // 4096x4096
    float* uact    = ws + o; o += (size_t)NTOK * D_INNER;        // 4096x2048
    float* dbc     = ws + o; o += (size_t)NTOK * DBC_LD;         // 4096x128 (padded)
    float* delta   = ws + o; o += (size_t)NTOK * D_INNER;        // 4096x2048
    float* ybuf    = ws + o; o += (size_t)NTOK * D_INNER;        // 4096x2048
    float* wxp_pad = ws + o;                                     // 2048x128

    // 0) pad W_xproj to 128 cols (guard-free GEMMs everywhere)
    pad_wxproj_kernel<<<(D_INNER * DBC_LD) / 256, 256, 0, stream>>>(W_xprj, wxp_pad);

    // 1) LayerNorm
    ln_kernel<<<NTOK, 256, 0, stream>>>(x, ln_w, ln_b, normed);

    // 2) xz = normed @ W_in   (M=4096, N=4096, K=1024)
    {
        dim3 g((2 * D_INNER) / 128, NTOK / 128);
        wmma_gemm<0><<<g, 256, 0, stream>>>(normed, D_MODEL, W_in, 2 * D_INNER,
                                            xz, 2 * D_INNER, NTOK, 2 * D_INNER, D_MODEL,
                                            nullptr, nullptr, 0);
    }

    // 3) depthwise causal conv + SiLU -> uact
    conv_silu_kernel<<<(NTOK * D_INNER) / 256, 256, 0, stream>>>(xz, conv_w, conv_b, uact);

    // 4) dbc = uact @ W_xproj_padded  (M=4096, N=128, K=2048)
    {
        dim3 g(DBC_LD / 128, NTOK / 128);
        wmma_gemm<0><<<g, 256, 0, stream>>>(uact, D_INNER, wxp_pad, DBC_LD,
                                            dbc, DBC_LD, NTOK, DBC_LD, D_INNER,
                                            nullptr, nullptr, 0);
    }

    // 5) delta = softplus(dbc[:, :64] @ W_dt + b_dt)  (M=4096, N=2048, K=64)
    {
        dim3 g(D_INNER / 128, NTOK / 128);
        wmma_gemm<1><<<g, 256, 0, stream>>>(dbc, DBC_LD, W_dt, D_INNER,
                                            delta, D_INNER, NTOK, D_INNER, DT_RANK,
                                            b_dt, nullptr, 0);
    }

    // 6) selective scan + gating -> ybuf
    scan_kernel<<<(BATCH * D_INNER) / 256, 256, 0, stream>>>(delta, uact, dbc, xz,
                                                             A_log, Dp, ybuf);

    // 7) out = x + ybuf @ W_out   (M=4096, N=1024, K=2048) with residual fused
    {
        dim3 g(D_MODEL / 128, NTOK / 128);
        wmma_gemm<2><<<g, 256, 0, stream>>>(ybuf, D_INNER, W_out, D_MODEL,
                                            out, D_MODEL, NTOK, D_MODEL, D_INNER,
                                            nullptr, x, D_MODEL);
    }
}
